// DotProductAttention_47175920779471
// MI455X (gfx1250) — matrix operand order repaired and checked by reading, not compiled, not run on hardware
//
#include <hip/hip_runtime.h>

typedef __attribute__((ext_vector_type(16))) _Float16 v16h;
typedef __attribute__((ext_vector_type(8)))  _Float16 v8h;
typedef __attribute__((ext_vector_type(4)))  _Float16 v4h;
typedef __attribute__((ext_vector_type(2)))  _Float16 v2h;
typedef __attribute__((ext_vector_type(8)))  float    v8f;

#define B_     16
#define LQ     2048
#define LK     2048
#define DH     64
#define DV     64
#define NEGVAL (-1000000000.0f)
#define LOG2E  1.44269504088896340736f

#define QTILE  128   // queries per workgroup (8 waves x 16 query-columns)
#define KBLK   64    // keys per mainloop iteration (16 WMMAs / iter / wave)

// A-operand fetch from LDS: halfs 0..7 at p, halfs 8..15 at p+8 (both 16B aligned)
__device__ __forceinline__ v16h ldA(const _Float16* p) {
    v8h a = *(const v8h*)p;
    v8h b = *(const v8h*)(p + 8);
    return __builtin_shufflevector(a, b, 0,1,2,3,4,5,6,7,8,9,10,11,12,13,14,15);
}

__device__ __forceinline__ int pack2(float a, float b) {
    union { v2h h; int i; } u;
    u.h[0] = (_Float16)a; u.h[1] = (_Float16)b;   // v_cvt_pk_f16_f32
    return u.i;
}

// P^T B-operand for one 32-key chunk: each lane sends exactly the tile its
// xor-16 partner needs (4 shuffles instead of 8).
__device__ __forceinline__ v16h buildB(const v8f& sa, const v8f& sb, int hi) {
    int t0d[4], t1d[4], snd[4], rcv[4];
    #pragma unroll
    for (int i = 0; i < 4; ++i) {
        t0d[i] = pack2(sa[2*i], sa[2*i+1]);     // keys 8*hi + {2i,2i+1}
        t1d[i] = pack2(sb[2*i], sb[2*i+1]);     // keys 16 + 8*hi + {2i,2i+1}
        snd[i] = hi ? t0d[i] : t1d[i];          // what the partner needs
    }
    #pragma unroll
    for (int i = 0; i < 4; ++i) rcv[i] = __shfl_xor(snd[i], 16, 32);
    union { int i[8]; v16h h; } pb;
    #pragma unroll
    for (int i = 0; i < 4; ++i) {
        pb.i[i]     = hi ? rcv[i] : t0d[i];     // kk 16*hi + 0..7
        pb.i[4 + i] = hi ? t1d[i] : rcv[i];     // kk 16*hi + 8..15
    }
    return pb.h;
}

__device__ __forceinline__ void wait_async() {
    asm volatile("s_wait_asynccnt 0x0" ::: "memory");
}

#if __has_builtin(__builtin_amdgcn_tensor_load_to_lds)
#define HAVE_TDM 1
typedef __attribute__((ext_vector_type(4))) unsigned int v4u;
typedef __attribute__((ext_vector_type(8))) int          v8i;
typedef __attribute__((ext_vector_type(4))) int          v4i;

// One TDM descriptor load: 64x64 f16 tile -> LDS, with HW row padding of
// 4 DWORDs (16B) after every 32 DWORDs (128B row) to match [..][64+8] layout.
__device__ __forceinline__ void tdm_tile_load(unsigned lds_off, const void* gsrc,
                                              unsigned tdim0, unsigned tdim1,
                                              unsigned long long d0stride) {
    unsigned long long ga = (unsigned long long)(uintptr_t)gsrc;
    v4u g0;
    g0[0] = 1u;                                                   // count=1 (valid)
    g0[1] = lds_off;                                              // LDS byte address
    g0[2] = (unsigned)ga;                                         // global_addr[31:0]
    g0[3] = (unsigned)((ga >> 32) & 0x01FFFFFFu) | 0x80000000u;   // addr[56:32] | type=2
    v8i g1;
    g1[0] = (int)(0x10000u         // data_size = 1 (2 bytes)
                | 0x100000u        // pad_enable
                | (4u << 22)       // pad_interval: 32 DWORDs between pads
                | (3u << 25));     // pad_amount: 4 DWORDs
    g1[1] = (int)(tdim0 << 16);                                   // tensor_dim0 lo
    g1[2] = (int)((tdim0 >> 16) | (tdim1 << 16));                 // dim0 hi | dim1 lo
    g1[3] = (int)((tdim1 >> 16) | (64u << 16));                   // dim1 hi | tile_dim0=64
    g1[4] = 64;                                                   // tile_dim1=64, tile_dim2=0
    g1[5] = (int)(unsigned)d0stride;                              // dim0_stride[31:0]
    g1[6] = (int)(unsigned)(d0stride >> 32);                      // dim0_stride[47:32]
    g1[7] = 0;
    v4i zero4 = {0, 0, 0, 0};
    v8i zero8 = {0, 0, 0, 0, 0, 0, 0, 0};
    __builtin_amdgcn_tensor_load_to_lds(g0, g1, zero4, zero4, zero8, 0);
}
#endif

// -------- pre-pass: k -> f16 (row major), v -> f16 transposed [b][dv][key] --------
__global__ __launch_bounds__(256)
void cvt_kv_kernel(const float* __restrict__ k, const float* __restrict__ v,
                   void* khv, void* vtv) {
    _Float16* kh = (_Float16*)khv;
    _Float16* vt = (_Float16*)vtv;
    size_t i = (size_t)blockIdx.x * 256 + threadIdx.x;   // 0 .. B*LK*DH-1
    kh[i] = (_Float16)k[i];
    size_t b   = i / ((size_t)DV * LK);
    size_t r   = i % ((size_t)DV * LK);
    size_t dv  = r / LK;
    size_t key = r % LK;
    vt[i] = (_Float16)v[(b * LK + key) * DV + dv];
}

// -------- main attention kernel --------
template<bool PRE16>
__global__ __launch_bounds__(256)
void DotProductAttention_47175920779471_kernel(const float* __restrict__ q,
                                               const float* __restrict__ kf,
                                               const float* __restrict__ vf,
                                               const void*  kh16,
                                               const void*  vt16,
                                               const int*   __restrict__ vlen,
                                               float* __restrict__ out)
{
    __shared__ __align__(16) _Float16 klds[2][KBLK][DH + 8];    // [buf][key][d]
    __shared__ __align__(16) _Float16 vlds[2][DV][KBLK + 8];    // [buf][dv][key]

    const int tid  = threadIdx.x;
    const int lane = tid & 31;
    const int n    = lane & 15;   // query column owned by this lane
    const int hi   = lane >> 4;
    const int wv   = tid >> 5;

    const int b  = blockIdx.y;
    const int q0 = blockIdx.x * QTILE + wv * 16;

    const float*    qb  = q + (size_t)b * LQ * DH;
    const float*    kfb = PRE16 ? nullptr : kf + (size_t)b * LK * DH;
    const float*    vfb = PRE16 ? nullptr : vf + (size_t)b * LK * DV;
    const _Float16* khb = PRE16 ? (const _Float16*)kh16 + (size_t)b * LK * DH : nullptr;
    const _Float16* vtb = PRE16 ? (const _Float16*)vt16 + (size_t)b * DV * LK : nullptr;

    // ---- Q^T as WMMA B operand, pre-scaled by log2(e) (softmax in exp2 domain) ----
    v16h qB0, qB1;
    {
        const float* qrow = qb + (size_t)(q0 + n) * DH + 16 * hi;
        #pragma unroll
        for (int h = 0; h < 16; ++h) {
            qB0[h] = (_Float16)(qrow[h]      * LOG2E);
            qB1[h] = (_Float16)(qrow[32 + h] * LOG2E);
        }
    }

    const int vl = vlen[(size_t)b * LQ + q0 + n];
    int vmax = vl, vmin = vl;
    #pragma unroll
    for (int off = 1; off <= 8; off <<= 1) {
        vmax = max(vmax, __shfl_xor(vmax, off, 32));
        vmin = min(vmin, __shfl_xor(vmin, off, 32));
    }

    v8f o[4] = {v8f{}, v8f{}, v8f{}, v8f{}};
    float rmax = -3.0e38f, rsum = 0.0f;

    // ---- staging: fill buffer `bf` with key block at kb ----
    auto stage = [&](int bf, int kb) {
        if constexpr (PRE16) {
#ifdef HAVE_TDM
            // Tensor Data Mover: wave 0 issues two descriptor loads for the
            // whole workgroup (EXEC ignored; tracked by TENSORcnt).
            if (wv == 0) {
                tdm_tile_load((unsigned)(uintptr_t)&klds[bf][0][0],
                              khb + (size_t)kb * DH, DH, LK, DH);
                tdm_tile_load((unsigned)(uintptr_t)&vlds[bf][0][0],
                              vtb + kb, LK, DV, LK);
            }
#else
            // per-lane async copies: 16B per lane-op, ASYNCcnt tracked
            #pragma unroll
            for (int it = 0; it < 2; ++it) {
                int c   = tid + it * 256;
                int row = c >> 3;
                int col = (c & 7) * 8;
                unsigned kdst = (unsigned)(uintptr_t)&klds[bf][row][col];
                const _Float16* ksrc = khb + (size_t)(kb + row) * DH + col;
                asm volatile("global_load_async_to_lds_b128 %0, %1, off"
                             :: "v"(kdst), "v"(ksrc) : "memory");
                unsigned vdst = (unsigned)(uintptr_t)&vlds[bf][row][col];
                const _Float16* vsrc = vtb + (size_t)row * LK + kb + col;
                asm volatile("global_load_async_to_lds_b128 %0, %1, off"
                             :: "v"(vdst), "v"(vsrc) : "memory");
            }
#endif
        } else {
            // fallback: load f32, convert, store (K packed b64; V transposed b16)
            #pragma unroll
            for (int it = 0; it < 4; ++it) {
                int idx = tid + it * 256;
                int kr  = idx >> 4;
                int dc  = (idx & 15) * 4;
                const float4 f = *(const float4*)(kfb + (size_t)(kb + kr) * DH + dc);
                v4h kp; kp[0] = (_Float16)f.x; kp[1] = (_Float16)f.y;
                        kp[2] = (_Float16)f.z; kp[3] = (_Float16)f.w;
                *(v4h*)&klds[bf][kr][dc] = kp;
                const float4 g = *(const float4*)(vfb + (size_t)(kb + kr) * DV + dc);
                vlds[bf][dc + 0][kr] = (_Float16)g.x;
                vlds[bf][dc + 1][kr] = (_Float16)g.y;
                vlds[bf][dc + 2][kr] = (_Float16)g.z;
                vlds[bf][dc + 3][kr] = (_Float16)g.w;
            }
        }
    };

    auto stage_wait = [&]() {
        if constexpr (PRE16) {
#ifdef HAVE_TDM
            if (wv == 0) __builtin_amdgcn_s_wait_tensorcnt(0);
#else
            wait_async();
#endif
        }
    };

    // prologue: fill buffer 0
    stage(0, 0);
    stage_wait();
    __syncthreads();

    int cur = 0;
    for (int kb0 = 0; kb0 < LK; kb0 += KBLK) {
        // issue next block's staging into the other buffer (overlaps with compute)
        if (kb0 + KBLK < LK) stage(cur ^ 1, kb0 + KBLK);

        // wave-uniform skip once past every query's valid length (exp underflows to 0)
        if (!(vmin > 0 && kb0 >= vmax)) {
            const _Float16 (*kl)[DH + 8]   = klds[cur];
            const _Float16 (*vt)[KBLK + 8] = vlds[cur];

            // ---- S^T = K @ Q^T : 4 C tiles of 16 keys ----
            v8f s[4];
            #pragma unroll
            for (int T = 0; T < 4; ++T) {
                v16h a0 = ldA(&kl[16 * T + n][16 * hi]);
                v16h a1 = ldA(&kl[16 * T + n][32 + 16 * hi]);
                v8f acc = {};
                acc = __builtin_amdgcn_wmma_f32_16x16x32_f16(false, a0, false, qB0, (short)0, acc, false, false);
                acc = __builtin_amdgcn_wmma_f32_16x16x32_f16(false, a1, false, qB1, (short)0, acc, false, false);
                s[T] = acc;
            }

            // ---- mask: compares against inline constants via per-lane threshold ----
            if (kb0 + KBLK > vmin) {
                const int tv = vl - kb0 - 8 * hi;   // key < vl  <=>  16T+j < tv
                #pragma unroll
                for (int T = 0; T < 4; ++T)
                    #pragma unroll
                    for (int j = 0; j < 8; ++j)
                        s[T][j] = (16 * T + j < tv) ? s[T][j] : NEGVAL;
            }

            // ---- online softmax: in-lane max tree + ONE xor-16 exchange ----
            float bm = s[0][0];
            #pragma unroll
            for (int T = 0; T < 4; ++T)
                #pragma unroll
                for (int j = 0; j < 8; ++j) bm = fmaxf(bm, s[T][j]);
            bm = fmaxf(bm, __shfl_xor(bm, 16, 32));
            float nm = fmaxf(rmax, bm);

            // deferred rescale: only when some query's max actually moved
            if (__ballot(nm > rmax)) {
                float corr = __builtin_amdgcn_exp2f(rmax - nm);
                rsum *= corr;
                #pragma unroll
                for (int t = 0; t < 4; ++t)
                    #pragma unroll
                    for (int j = 0; j < 8; ++j) o[t][j] *= corr;
            }
            rmax = nm;

            // ---- probs in-place (exp2 domain) ----
            #pragma unroll
            for (int T = 0; T < 4; ++T)
                #pragma unroll
                for (int j = 0; j < 8; ++j)
                    s[T][j] = __builtin_amdgcn_exp2f(s[T][j] - nm);

            // balanced partial-sum tree (vector pk adds + short horizontal chain)
            v8f a4 = (s[0] + s[1]) + (s[2] + s[3]);
            float part = ((a4[0] + a4[4]) + (a4[1] + a4[5]))
                       + ((a4[2] + a4[6]) + (a4[3] + a4[7]));
            rsum += part;

            // ---- P^T B operands (two 32-key chunks) ----
            v16h pb0 = buildB(s[0], s[1], hi);
            v16h pb1 = buildB(s[2], s[3], hi);

            // ---- O^T += V^T @ P^T : 4 dv tiles x 2 kk chunks ----
            #pragma unroll
            for (int t = 0; t < 4; ++t) {
                v16h aV0 = ldA(&vt[16 * t + n][16 * hi]);
                v16h aV1 = ldA(&vt[16 * t + n][32 + 16 * hi]);
                o[t] = __builtin_amdgcn_wmma_f32_16x16x32_f16(false, aV0, false, pb0, (short)0, o[t], false, false);
                o[t] = __builtin_amdgcn_wmma_f32_16x16x32_f16(false, aV1, false, pb1, (short)0, o[t], false, false);
            }
        }

        stage_wait();
        __syncthreads();          // next buffer staged; everyone done with cur
        cur ^= 1;
    }

    // ---- combine partner partial sums, normalize, store O^T ----
    rsum += __shfl_xor(rsum, 16, 32);
    const float inv = 1.0f / rsum;
    float* ocol = out + ((size_t)b * LQ + q0 + n) * DV;
    #pragma unroll
    for (int t = 0; t < 4; ++t)
        #pragma unroll
        for (int j = 0; j < 8; ++j)
            ocol[16 * t + 8 * hi + j] = o[t][j] * inv;
}

extern "C" void kernel_launch(void* const* d_in, const int* in_sizes, int n_in,
                              void* d_out, int out_size, void* d_ws, size_t ws_size,
                              hipStream_t stream) {
    const float* q  = (const float*)d_in[0];
    const float* k  = (const float*)d_in[1];
    const float* v  = (const float*)d_in[2];
    const int*   vl = (const int*)d_in[3];
    float* o        = (float*)d_out;

    const size_t khBytes = (size_t)B_ * LK * DH * sizeof(short);   // 4 MB
    const size_t vtBytes = (size_t)B_ * DV * LK * sizeof(short);   // 4 MB
    dim3 grid(LQ / QTILE, B_);

    if (ws_size >= khBytes + vtBytes) {
        void* kh = d_ws;
        void* vt = (char*)d_ws + khBytes;
        size_t nElem = (size_t)B_ * LK * DH;
        cvt_kv_kernel<<<(unsigned)(nElem / 256), 256, 0, stream>>>(k, v, kh, vt);
        DotProductAttention_47175920779471_kernel<true>
            <<<grid, 256, 0, stream>>>(q, nullptr, nullptr, kh, vt, vl, o);
    } else {
        DotProductAttention_47175920779471_kernel<false>
            <<<grid, 256, 0, stream>>>(q, k, v, nullptr, nullptr, vl, o);
    }
}
